// DCN2_OP_52218212385394
// MI455X (gfx1250) — matrix operand order repaired and checked by reading, not compiled, not run on hardware
//
#include <hip/hip_runtime.h>

// ---------------- problem constants ----------------
#define B_    8
#define H_    32
#define W_    32
#define P_    1024          // H*W
#define C_    384
#define G_    4
#define CG_   96            // C/G
#define KK_   9
#define OFFC  72            // G*2*KK
#define OFFCP 80            // padded to 5 N-tiles of 16
#define KOFF  (KK_*C_)      // 3456
#define KDCN  (KK_*C_)      // 3456

typedef __attribute__((ext_vector_type(8)))  __bf16 bf16x8;
typedef __attribute__((ext_vector_type(16))) __bf16 bf16x16;
typedef __attribute__((ext_vector_type(8)))  float  f32x8;

union AFrag { bf16x16 v; bf16x8 h[2]; uint4 q[2]; };
union BFrag { bf16x16 v; uint4 q[2]; };

static __device__ __forceinline__ f32x8 wmma_bf16(bf16x16 a, bf16x16 b, f32x8 c) {
  // (neg_a, A, neg_b, B, c_mod, C, reuse_a, reuse_b)
  return __builtin_amdgcn_wmma_f32_16x16x32_bf16(false, a, false, b, (short)0, c, false, false);
}

static __device__ __forceinline__ f32x8 zero8() {
  f32x8 z;
#pragma unroll
  for (int i = 0; i < 8; i++) z[i] = 0.0f;
  return z;
}

// native HW conversions (single cvt ops on gfx1250)
static __device__ __forceinline__ unsigned short f2bf(float f) {
  union { __bf16 b; unsigned short u; } cv;
  cv.b = (__bf16)f;
  return cv.u;
}
static __device__ __forceinline__ void ld8f(const unsigned short* p, float* f) {
  bf16x8 v = *(const bf16x8*)p;
#pragma unroll
  for (int i = 0; i < 8; i++) f[i] = (float)v[i];
}

// ---------------- kernel 0: precompute bf16 copies / transposed weights / BN affine ----------------
__global__ void k_prep(const float* __restrict__ x, const float* __restrict__ w_off,
                       const float* __restrict__ w_dcn, const float* __restrict__ gamma,
                       const float* __restrict__ beta, const float* __restrict__ mean,
                       const float* __restrict__ var, const float* __restrict__ w2,
                       unsigned short* __restrict__ xbf, unsigned short* __restrict__ wofft,
                       unsigned short* __restrict__ wdcnt, unsigned short* __restrict__ w2t,
                       float* __restrict__ bnscale, float* __restrict__ bnshift) {
  long i = (long)blockIdx.x * blockDim.x + threadIdx.x;
  const long NX  = (long)B_ * P_ * C_;      // 3145728
  const long NWO = (long)OFFCP * KOFF;      // 276480
  const long NWD = (long)C_ * KDCN;         // 1327104
  const long NW2 = (long)C_ * C_;           // 147456
  if (i < NX) { xbf[i] = f2bf(x[i]); return; }
  i -= NX;
  if (i < NWO) {               // wofft[oc][n*C + c] = w_off[oc][c][n], zero-pad oc>=72
    int oc = (int)(i / KOFF); int k = (int)(i % KOFF); int n = k / C_, c = k % C_;
    float v = (oc < OFFC) ? w_off[((long)oc * C_ + c) * KK_ + n] : 0.0f;
    wofft[i] = f2bf(v); return;
  }
  i -= NWO;
  if (i < NWD) {               // wdcnt[o][kk*C + c] = w_dcn[o][c][kk]
    int o = (int)(i / KDCN); int k = (int)(i % KDCN); int kk = k / C_, c = k % C_;
    wdcnt[i] = f2bf(w_dcn[((long)o * C_ + c) * KK_ + kk]); return;
  }
  i -= NWD;
  if (i < NW2) { w2t[i] = f2bf(w2[i]); return; }   // w2t[o][c] = w2[o][c]
  i -= NW2;
  if (i < C_) {
    float sc = gamma[i] * rsqrtf(var[i] + 1e-5f);
    bnscale[i] = sc; bnshift[i] = beta[i] - mean[i] * sc;
  }
}

// ---------------- kernel 1: 3x3 offset conv as WMMA GEMM (M=64/WG, N=80, K=9*384) ----------------
__global__ __launch_bounds__(128) void k_offconv(const unsigned short* __restrict__ xbf,
                                                 const unsigned short* __restrict__ wofft,
                                                 const float* __restrict__ b_off,
                                                 float* __restrict__ offbuf) {
  const int b = blockIdx.y, ptile = blockIdx.x;
  const int tid = threadIdx.x, lane = tid & 31;
  const int lm = lane & 15, lk = lane >> 4;
  const int mt = tid >> 5;                  // 4 waves, one 16-row tile each
  const int p = ptile * 64 + mt * 16 + lm;  // this lane's A row
  const int yy = p >> 5, xx = p & 31;

  f32x8 acc[5];
#pragma unroll
  for (int nt = 0; nt < 5; nt++) acc[nt] = zero8();

  // per-lane invariant pieces of the B address
  const unsigned short* bbase0 = wofft + (long)lm * KOFF + lk * 8;

  for (int n = 0; n < 9; n++) {
    const int ny = yy + n / 3 - 1, nx = xx + n % 3 - 1;
    const bool valid = (ny >= 0) & (ny < H_) & (nx >= 0) & (nx < W_);
    const unsigned short* arow = xbf + ((long)b * P_ + (valid ? ny * W_ + nx : 0)) * C_ + lk * 8;
    const unsigned short* brow = bbase0 + n * C_ + lk * 16 - lk * 16; // keep simple
    for (int ck = 0; ck < 12; ck++) {
      const int c0 = ck * 32;
      AFrag a;
      if (valid) {
        a.q[0] = *(const uint4*)(arow + c0);
        a.q[1] = *(const uint4*)(arow + c0 + 16);
      } else {
        a.q[0] = make_uint4(0, 0, 0, 0); a.q[1] = make_uint4(0, 0, 0, 0);
      }
      // batch-load all B fragments, then issue the WMMA burst
      BFrag bb[5];
#pragma unroll
      for (int nt = 0; nt < 5; nt++) {
        const unsigned short* bp = bbase0 + (long)nt * 16 * KOFF + n * C_ + c0;
        bb[nt].q[0] = *(const uint4*)bp;
        bb[nt].q[1] = *(const uint4*)(bp + 16);
      }
#pragma unroll
      for (int nt = 0; nt < 5; nt++) acc[nt] = wmma_bf16(a.v, bb[nt].v, acc[nt]);
    }
  }
#pragma unroll
  for (int nt = 0; nt < 5; nt++) {
    const int oc = nt * 16 + lm;
    if (oc < OFFC) {
      const float bias = b_off[oc];
#pragma unroll
      for (int v = 0; v < 8; v++) {
        const int mm = mt * 16 + v + lk * 8;
        offbuf[((long)b * P_ + ptile * 64 + mm) * OFFC + oc] = acc[nt][v] + bias;
      }
    }
  }
}

// ---------------- kernel 2: fused bilinear sampling + DCN GEMM + BN + SiLU ----------------
__global__ __launch_bounds__(256) void k_dcn(const unsigned short* __restrict__ xbf,
                                             const unsigned short* __restrict__ wdcnt,
                                             const float* __restrict__ offbuf,
                                             const float* __restrict__ bnscale,
                                             const float* __restrict__ bnshift,
                                             unsigned short* __restrict__ ybf) {
  __shared__ __align__(16) unsigned short sA[64 * C_];   // 48 KB sampled tile (64 pos x 384 ch, bf16)
  const int b = blockIdx.y, ptile = blockIdx.x;
  const int tid = threadIdx.x, lane = tid & 31;
  const int lm = lane & 15, lk = lane >> 4;
  const int wv = tid >> 5, mt = wv & 3, nh = wv >> 2;    // 8 waves: 4 row-tiles x 2 col-halves

  f32x8 acc[12];
#pragma unroll
  for (int nt = 0; nt < 12; nt++) acc[nt] = zero8();

  // sampling mapping: one (position, group) per thread
  const int pl = tid >> 2, g = tid & 3;
  const int p = ptile * 64 + pl;
  const int yy = p >> 5, xx = p & 31;
  const unsigned short* xb = xbf + (long)b * P_ * C_;
  const float* offrow = offbuf + ((long)b * P_ + p) * OFFC + g * 18;

  // per-lane invariant B base: column (output channel) fixed per lane half
  const unsigned short* bcol = wdcnt + (long)(nh * 192 + lm) * KDCN + lk * 8;

  for (int kk = 0; kk < 9; kk++) {
    __syncthreads();
    { // ---- build sampled tile for tap kk ----
      const float dy = offrow[kk * 2 + 0], dx = offrow[kk * 2 + 1];
      const float py = dy + (float)(kk / 3 - 1) + (float)yy;
      const float px = dx + (float)(kk % 3 - 1) + (float)xx;
      const float fy = floorf(py), fx = floorf(px);
      const int iy0 = (int)fy, ix0 = (int)fx;
      const float wy1 = py - fy, wx1 = px - fx, wy0 = 1.f - wy1, wx0 = 1.f - wx1;
      const bool vy0 = (iy0 >= 0) & (iy0 < H_), vy1 = (iy0 >= -1) & (iy0 < H_ - 1);
      const bool vx0 = (ix0 >= 0) & (ix0 < W_), vx1 = (ix0 >= -1) & (ix0 < W_ - 1);
      const int cy0 = min(max(iy0, 0), H_ - 1), cy1 = min(max(iy0 + 1, 0), H_ - 1);
      const int cx0 = min(max(ix0, 0), W_ - 1), cx1 = min(max(ix0 + 1, 0), W_ - 1);
      const unsigned short* r00 = xb + (cy0 * W_ + cx0) * C_;
      const unsigned short* r01 = xb + (cy0 * W_ + cx1) * C_;
      const unsigned short* r10 = xb + (cy1 * W_ + cx0) * C_;
      const unsigned short* r11 = xb + (cy1 * W_ + cx1) * C_;
      const float w00 = (vy0 & vx0) ? wy0 * wx0 : 0.f;
      const float w01 = (vy0 & vx1) ? wy0 * wx1 : 0.f;
      const float w10 = (vy1 & vx0) ? wy1 * wx0 : 0.f;
      const float w11 = (vy1 & vx1) ? wy1 * wx1 : 0.f;
#pragma unroll
      for (int j = 0; j < 12; j++) {      // 96 channels of this group, 8 at a time
        const int c0 = g * CG_ + j * 8;
        float f00[8], f01[8], f10[8], f11[8];
        ld8f(r00 + c0, f00); ld8f(r01 + c0, f01);
        ld8f(r10 + c0, f10); ld8f(r11 + c0, f11);
        bf16x8 vv;
#pragma unroll
        for (int i = 0; i < 8; i++)
          vv[i] = (__bf16)(w00 * f00[i] + w01 * f01[i] + w10 * f10[i] + w11 * f11[i]);
        *(bf16x8*)&sA[pl * C_ + c0] = vv;   // single ds_store_b128
      }
    }
    __syncthreads();
    // ---- GEMM over this tap's 384 K values ----
    const unsigned short* bkk = bcol + (long)kk * C_;
    for (int ck = 0; ck < 12; ck++) {
      const int c0 = ck * 32;
      const unsigned short* ab = &sA[(mt * 16 + lm) * C_ + c0];
      AFrag a;
      a.q[0] = *(const uint4*)(ab + lk * 8);
      a.q[1] = *(const uint4*)(ab + 16 + lk * 8);
      // batch-load all 12 B fragments (24 x b128 clause), then WMMA burst
      BFrag bb[12];
#pragma unroll
      for (int nt = 0; nt < 12; nt++) {
        const unsigned short* bp = bkk + (long)nt * 16 * KDCN + c0;
        bb[nt].q[0] = *(const uint4*)bp;
        bb[nt].q[1] = *(const uint4*)(bp + 16);
      }
#pragma unroll
      for (int nt = 0; nt < 12; nt++) acc[nt] = wmma_bf16(a.v, bb[nt].v, acc[nt]);
    }
  }
  // ---- BN + SiLU, write bf16 activations ----
#pragma unroll
  for (int nt = 0; nt < 12; nt++) {
    const int c = nh * 192 + nt * 16 + lm;
    const float sc = bnscale[c], sh = bnshift[c];
#pragma unroll
    for (int v = 0; v < 8; v++) {
      const int mm = mt * 16 + v + lk * 8;
      const float val = acc[nt][v] * sc + sh;
      const float s = val * (1.0f / (1.0f + __expf(-val)));
      ybf[((long)b * P_ + ptile * 64 + mm) * C_ + c] = f2bf(s);
    }
  }
}

// ---------------- kernel 3: pointwise GEMM (M=8192, N=384, K=384) + bias -> f32 out ----------------
__global__ __launch_bounds__(256) void k_pw(const unsigned short* __restrict__ ybf,
                                            const unsigned short* __restrict__ w2t,
                                            const float* __restrict__ b2,
                                            float* __restrict__ out) {
  const int b = blockIdx.y, ptile = blockIdx.x;
  const int tid = threadIdx.x, lane = tid & 31;
  const int lm = lane & 15, lk = lane >> 4;
  const int wv = tid >> 5, mt = wv & 3, nh = wv >> 2;

  f32x8 acc[12];
#pragma unroll
  for (int nt = 0; nt < 12; nt++) acc[nt] = zero8();

  const unsigned short* arow = ybf + ((long)b * P_ + ptile * 64 + mt * 16 + lm) * C_ + lk * 8;
  const unsigned short* bcol = w2t + (long)(nh * 192 + lm) * C_ + lk * 8;
  for (int ck = 0; ck < 12; ck++) {
    const int c0 = ck * 32;
    AFrag a;
    a.q[0] = *(const uint4*)(arow + c0);
    a.q[1] = *(const uint4*)(arow + c0 + 16);
    BFrag bb[12];
#pragma unroll
    for (int nt = 0; nt < 12; nt++) {
      const unsigned short* bp = bcol + (long)nt * 16 * C_ + c0;
      bb[nt].q[0] = *(const uint4*)bp;
      bb[nt].q[1] = *(const uint4*)(bp + 16);
    }
#pragma unroll
    for (int nt = 0; nt < 12; nt++) acc[nt] = wmma_bf16(a.v, bb[nt].v, acc[nt]);
  }
#pragma unroll
  for (int nt = 0; nt < 12; nt++) {
    const int oc = nh * 192 + nt * 16 + lm;
    const float bias = b2[oc];
#pragma unroll
    for (int v = 0; v < 8; v++) {
      const int mm = mt * 16 + v + lk * 8;
      out[((long)b * P_ + ptile * 64 + mm) * C_ + oc] = acc[nt][v] + bias;
    }
  }
}

// ---------------- host ----------------
extern "C" void kernel_launch(void* const* d_in, const int* in_sizes, int n_in,
                              void* d_out, int out_size, void* d_ws, size_t ws_size,
                              hipStream_t stream) {
  (void)in_sizes; (void)n_in; (void)out_size; (void)ws_size;
  const float* x     = (const float*)d_in[0];
  const float* w_off = (const float*)d_in[1];
  const float* b_off = (const float*)d_in[2];
  const float* w_dcn = (const float*)d_in[3];
  const float* gamma = (const float*)d_in[4];
  const float* beta  = (const float*)d_in[5];
  const float* mean  = (const float*)d_in[6];
  const float* var   = (const float*)d_in[7];
  const float* w2    = (const float*)d_in[8];
  const float* b2    = (const float*)d_in[9];

  char* ws = (char*)d_ws;                       // all offsets 256B-aligned
  unsigned short* xbf   = (unsigned short*)(ws + 0);         // 6,291,456 B
  unsigned short* wofft = (unsigned short*)(ws + 6291456);   //   552,960 B
  unsigned short* wdcnt = (unsigned short*)(ws + 6844416);   // 2,654,208 B
  unsigned short* w2t   = (unsigned short*)(ws + 9498624);   //   294,912 B
  float* bnscale        = (float*)(ws + 9793536);            //     1,536 B
  float* bnshift        = (float*)(ws + 9795072);            //     1,536 B
  float* offbuf         = (float*)(ws + 9796608);            // 2,359,296 B
  unsigned short* ybf   = (unsigned short*)(ws + 12155904);  // 6,291,456 B
  float* out = (float*)d_out;

  const long total = (long)B_ * P_ * C_ + (long)OFFCP * KOFF + (long)C_ * KDCN + (long)C_ * C_ + C_;
  const int blocks = (int)((total + 255) / 256);
  k_prep<<<blocks, 256, 0, stream>>>(x, w_off, w_dcn, gamma, beta, mean, var, w2,
                                     xbf, wofft, wdcnt, w2t, bnscale, bnshift);
  dim3 grid(P_ / 64, B_);
  k_offconv<<<grid, 128, 0, stream>>>(xbf, wofft, b_off, offbuf);
  k_dcn<<<grid, 256, 0, stream>>>(xbf, wdcnt, offbuf, bnscale, bnshift, ybf);
  k_pw<<<grid, 256, 0, stream>>>(ybf, w2t, b2, out);
}
